// OracleSparseIDM_3083786518538
// MI455X (gfx1250) — matrix operand order repaired and checked by reading, not compiled, not run on hardware
//
#include <hip/hip_runtime.h>

typedef __attribute__((ext_vector_type(16))) _Float16 v16h;
typedef __attribute__((ext_vector_type(8)))  float    v8f;

#define AGENT_ID 10
#define BATCH    32768
#define PI_APPROX 3.14159f

// ---------------------------------------------------------------------------
// Kernel 1: agent search + feature extraction. One wave32 per sample.
// Writes f16 feature matrix feat[B][32] (cols 0..17 real, 18..31 zero),
// a layout the MLP kernel can consume as WMMA A-operands with contiguous loads.
// ---------------------------------------------------------------------------
__global__ __launch_bounds__(256) void feat_kernel(
    const int* __restrict__ frames,       // [2,B,32,32,3]
    const int* __restrict__ carried_col,  // [2,B,1]
    const int* __restrict__ carried_obj,  // [2,B,1]
    _Float16* __restrict__ feat)          // [B,32]
{
    const int lane = threadIdx.x & 31;
    const int wave = threadIdx.x >> 5;
    const int b    = blockIdx.x * 8 + wave;            // grid covers exactly B

    const size_t frameElems = (size_t)BATCH * 32 * 32 * 3;
    const int* curr = frames + (size_t)b * 32 * 32 * 3;
    const int* nxt  = curr + frameElems;

    // chunked row-major scan: expected ~1 iteration (P(cell==10) = 1/11)
    int found_idx = -1;
    for (int r = 0; r < 32; ++r) {
        int c0 = curr[(size_t)(r * 32 + lane) * 3];
        unsigned long long m = __ballot(c0 == AGENT_ID);
        if (m != 0ull) {                                // uniform across wave
            found_idx = r * 32 + __builtin_ctzll(m);
            break;
        }
    }
    int y = 16, x = 16;                                 // fallback H/2, W/2
    if (found_idx >= 0) { y = found_idx >> 5; x = found_idx & 31; }

    if (lane == 0) {
        const int cd = curr[(size_t)(y * 32 + x) * 3 + 2];
        const int nd = nxt [(size_t)(y * 32 + x) * 3 + 2];
        const int d4 = cd & 3;                          // cd >= 0
        const int dys[4] = {1, 0, -1, 0};
        const int dxs[4] = {0, 1, 0, -1};
        const int yy = y + dys[d4], xx = x + dxs[d4];
        const bool inb = (yy >= 0 && yy < 32 && xx >= 0 && xx < 32);

        float f[18];
        const int* cc = curr + (size_t)(y * 32 + x) * 3;
        f[0] = (float)cc[0]; f[1] = (float)cc[1]; f[2] = (float)cc[2];
        if (inb) {
            const int* cf = curr + (size_t)(yy * 32 + xx) * 3;
            f[3] = (float)cf[0]; f[4] = (float)cf[1]; f[5] = (float)cf[2];
        } else { f[3] = f[4] = f[5] = 0.f; }
        const int* nc = nxt + (size_t)(y * 32 + x) * 3;
        f[6] = (float)nc[0]; f[7] = (float)nc[1]; f[8] = (float)nc[2];
        if (inb) {
            const int* nf = nxt + (size_t)(yy * 32 + xx) * 3;
            f[9] = (float)nf[0]; f[10] = (float)nf[1]; f[11] = (float)nf[2];
        } else { f[9] = f[10] = f[11] = 0.f; }
        f[12] = (float)carried_col[b];
        f[13] = (float)carried_obj[b];
        f[14] = (float)carried_col[BATCH + b];
        f[15] = (float)carried_obj[BATCH + b];
        const int delta = ((nd - cd) % 4 + 4) % 4;      // python-mod semantics
        const float ang = (float)delta * (2.0f * PI_APPROX / 4.0f);
        f[16] = __sinf(ang);
        f[17] = __cosf(ang);

        _Float16* dst = feat + (size_t)b * 32;
#pragma unroll
        for (int i = 0; i < 18; ++i) dst[i] = (_Float16)f[i];
#pragma unroll
        for (int i = 18; i < 32; ++i) dst[i] = (_Float16)0.f;
    }
}

// ---------------------------------------------------------------------------
// Kernel 2: 18->64->64->8 MLP via v_wmma_f32_16x16x32_f16.
// One wave per 16-row tile of B. Weights pre-gathered into B-operand layout,
// hidden activations staged through per-wave LDS to rebuild A-operand layout.
// ---------------------------------------------------------------------------
union V16U {
    v16h v;
    uint4 u[2];
    _Float16 h[16];
};

__global__ __launch_bounds__(256) void mlp_kernel(
    const _Float16* __restrict__ feat,  // [B,32] f16
    const float* __restrict__ W1, const float* __restrict__ b1,   // [18,64],[64]
    const float* __restrict__ W2, const float* __restrict__ b2,   // [64,64],[64]
    const float* __restrict__ W3, const float* __restrict__ b3,   // [64,8],[8]
    float* __restrict__ out)            // [B,8] f32
{
    __shared__ __align__(16) _Float16 hbuf[8][16][64];  // per-wave staging, 16 KB

    const int lane = threadIdx.x & 31;
    const int wave = threadIdx.x >> 5;
    const int hi   = lane >> 4;     // lane-group: selects K half
    const int lo   = lane & 15;     // B-layout: N column / C-layout: N column

    // ---- gather weights into WMMA B-layout registers (f32 -> f16) ----
    V16U w1b[4], w2b[2][4], w3b[2];
#pragma unroll
    for (int n = 0; n < 4; ++n)
#pragma unroll
        for (int j = 0; j < 16; ++j) {
            const int k = hi * 8 + j + (j >= 8 ? 8 : 0);     // K index
            w1b[n].h[j] = (k < 18) ? (_Float16)W1[k * 64 + n * 16 + lo]
                                   : (_Float16)0.f;          // K-pad 18->32
        }
#pragma unroll
    for (int kb = 0; kb < 2; ++kb)
#pragma unroll
        for (int n = 0; n < 4; ++n)
#pragma unroll
            for (int j = 0; j < 16; ++j) {
                const int k = kb * 32 + hi * 8 + j + (j >= 8 ? 8 : 0);
                w2b[kb][n].h[j] = (_Float16)W2[k * 64 + n * 16 + lo];
            }
#pragma unroll
    for (int kb = 0; kb < 2; ++kb)
#pragma unroll
        for (int j = 0; j < 16; ++j) {
            const int k = kb * 32 + hi * 8 + j + (j >= 8 ? 8 : 0);
            w3b[kb].h[j] = (lo < 8) ? (_Float16)W3[k * 8 + lo]
                                    : (_Float16)0.f;         // N-pad 8->16
        }

    float bias1[4], bias2[4];
#pragma unroll
    for (int n = 0; n < 4; ++n) { bias1[n] = b1[n * 16 + lo]; bias2[n] = b2[n * 16 + lo]; }
    const float bias3 = (lo < 8) ? b3[lo] : 0.f;

    const int tile = blockIdx.x * 8 + wave;  // 256 blocks * 8 waves = 2048 tiles
    const int row0 = tile * 16;

    // ---- A0: feature tile [16 x 32] f16, straight from global in A layout ----
    V16U a0;
    {
        const uint4* p = reinterpret_cast<const uint4*>(feat) + (size_t)(row0 + lo) * 4 + hi;
        a0.u[0] = p[0];     // K 0..7   (lanes>=16: K 8..15)
        a0.u[1] = p[2];     // K 16..23 (lanes>=16: K 24..31)
    }

    // ---- layer 1: H1 = relu(X @ W1 + b1), K=32 (padded), N=64 ----
#pragma unroll
    for (int n = 0; n < 4; ++n) {
        v8f c;
#pragma unroll
        for (int r = 0; r < 8; ++r) c[r] = bias1[n];
        c = __builtin_amdgcn_wmma_f32_16x16x32_f16(false, a0.v, false, w1b[n].v,
                                                   (short)0, c, false, false);
#pragma unroll
        for (int r = 0; r < 8; ++r) {
            const float v = c[r] > 0.f ? c[r] : 0.f;
            hbuf[wave][r + hi * 8][n * 16 + lo] = (_Float16)v;   // D -> [M][N]
        }
    }
    __syncthreads();

    // rebuild A layout for K=64 (two A tiles) from LDS
    V16U a2[2];
#pragma unroll
    for (int t = 0; t < 2; ++t) {
        const uint4* lp = reinterpret_cast<const uint4*>(&hbuf[wave][lo][t * 32 + hi * 8]);
        a2[t].u[0] = lp[0];
        a2[t].u[1] = lp[2];
    }
    __syncthreads();

    // ---- layer 2: H2 = relu(H1 @ W2 + b2), K=64, N=64 ----
#pragma unroll
    for (int n = 0; n < 4; ++n) {
        v8f c;
#pragma unroll
        for (int r = 0; r < 8; ++r) c[r] = bias2[n];
        c = __builtin_amdgcn_wmma_f32_16x16x32_f16(false, a2[0].v, false, w2b[0][n].v,
                                                   (short)0, c, false, false);
        c = __builtin_amdgcn_wmma_f32_16x16x32_f16(false, a2[1].v, false, w2b[1][n].v,
                                                   (short)0, c, false, false);
#pragma unroll
        for (int r = 0; r < 8; ++r) {
            const float v = c[r] > 0.f ? c[r] : 0.f;
            hbuf[wave][r + hi * 8][n * 16 + lo] = (_Float16)v;
        }
    }
    __syncthreads();

    V16U a3[2];
#pragma unroll
    for (int t = 0; t < 2; ++t) {
        const uint4* lp = reinterpret_cast<const uint4*>(&hbuf[wave][lo][t * 32 + hi * 8]);
        a3[t].u[0] = lp[0];
        a3[t].u[1] = lp[2];
    }
    __syncthreads();

    // ---- layer 3: out = H2 @ W3 + b3, K=64, N=16 (cols 8..15 padded) ----
    v8f c3;
#pragma unroll
    for (int r = 0; r < 8; ++r) c3[r] = bias3;
    c3 = __builtin_amdgcn_wmma_f32_16x16x32_f16(false, a3[0].v, false, w3b[0].v,
                                                (short)0, c3, false, false);
    c3 = __builtin_amdgcn_wmma_f32_16x16x32_f16(false, a3[1].v, false, w3b[1].v,
                                                (short)0, c3, false, false);
    if (lo < 8) {
#pragma unroll
        for (int r = 0; r < 8; ++r)
            out[(size_t)(row0 + r + hi * 8) * 8 + lo] = c3[r];
    }
}

// ---------------------------------------------------------------------------
extern "C" void kernel_launch(void* const* d_in, const int* in_sizes, int n_in,
                              void* d_out, int out_size, void* d_ws, size_t ws_size,
                              hipStream_t stream) {
    const int*   frames = (const int*)d_in[0];
    const int*   ccol   = (const int*)d_in[1];
    const int*   cobj   = (const int*)d_in[2];
    const float* W1     = (const float*)d_in[3];
    const float* b1     = (const float*)d_in[4];
    const float* W2     = (const float*)d_in[5];
    const float* b2     = (const float*)d_in[6];
    const float* W3     = (const float*)d_in[7];
    const float* b3     = (const float*)d_in[8];
    float*       out    = (float*)d_out;
    _Float16*    feat   = (_Float16*)d_ws;          // B*32*2 = 2 MB scratch

    // 32768 samples, one wave32 per sample, 8 waves per block
    feat_kernel<<<BATCH / 8, 256, 0, stream>>>(frames, ccol, cobj, feat);
    // 2048 row-tiles of 16, one wave per tile, 8 waves per block
    mlp_kernel<<<2048 / 8, 256, 0, stream>>>(feat, W1, b1, W2, b2, W3, b3, out);
}
